// PureSSMCharLM_87041807220867
// MI455X (gfx1250) — matrix operand order repaired and checked by reading, not compiled, not run on hardware
//
#include <hip/hip_runtime.h>
#include <math.h>

// ---------------------------------------------------------------------------
// Problem constants (match reference)
// ---------------------------------------------------------------------------
#define BB     32
#define TT     2048
#define EMB    128
#define HH     512
#define LL     6
#define VOCAB  256
#define ROWS   (BB * TT)          // 65536
#define CS     128                // SSM chunk size
#define NCH    (TT / CS)          // 16 chunks

// ---------------------------------------------------------------------------
// WMMA fragment types (gfx1250, wave32)
// ---------------------------------------------------------------------------
typedef __attribute__((ext_vector_type(16))) __bf16          v16bf;
typedef __attribute__((ext_vector_type(8)))  float           v8f;
typedef __attribute__((ext_vector_type(8)))  unsigned short  u16x8;

union BfFrag {
    v16bf bf;
    u16x8 h[2];
};

// round-to-nearest-even f32 -> bf16
__device__ __forceinline__ unsigned short f2bf(float f) {
    unsigned int u = __float_as_uint(f);
    unsigned int r = u + 0x7FFFu + ((u >> 16) & 1u);
    return (unsigned short)(r >> 16);
}

// ---------------------------------------------------------------------------
// WMMA GEMM:  out[rows x N] = A_bf16[rows x K] * WT_bf16[N x K]^T + bias
// Per wave: 32 rows x 64 cols (8 accum tiles, 2 A frags, 4 B frags per K step;
// each B fragment feeds 2 WMMAs). Block = 256 thr = 8 waves -> 256 x 64 tile.
// Grid: (rows/256, N/64). rows%256==0, K%32==0, N%64==0 for all uses here.
//
// A layout (16-bit 16x32): lane half=l>>4, m=l&15:
//   elems 0..7 : K = k0 +      half*8 + e   (contiguous 16B)
//   elems 8..15: K = k0 + 16 + half*8 + e   (contiguous 16B)
// B layout (16-bit 32x16): n=l&15, khalf=l>>4 (same K order as A):
//   elems 0..7 : K = k0 + khalf*8 + e, elems 8..15: K = k0 + 16 + khalf*8 + e (row n of WT)
// C/D layout (f32 16x16): col = l&15, row = (l>>4)*8 + j
// ---------------------------------------------------------------------------
__global__ __launch_bounds__(256) void k_gemm_bf16(
    const unsigned short* __restrict__ A,
    const unsigned short* __restrict__ WT,
    const float* __restrict__ bias,            // may be nullptr
    float* __restrict__ out,
    unsigned short* __restrict__ out_bf,       // optional bf16 mirror (may be nullptr)
    int rows, int K, int N)
{
    const int wave = threadIdx.x >> 5;
    const int lane = threadIdx.x & 31;
    const int m    = lane & 15;
    const int half = lane >> 4;

    const int rowBase = blockIdx.x * 256 + wave * 32;
    const int colBase = blockIdx.y * 64;

    v8f acc[8] = {};   // [r*4 + t], r = row half (16 rows), t = col tile

    const unsigned short* __restrict__ ar0 = A + (size_t)(rowBase + m) * K + half * 8;
    const unsigned short* __restrict__ ar1 = ar0 + (size_t)16 * K;
    const unsigned short* __restrict__ w0 = WT + (size_t)(colBase + 0  + m) * K + half * 8;
    const unsigned short* __restrict__ w1 = WT + (size_t)(colBase + 16 + m) * K + half * 8;
    const unsigned short* __restrict__ w2 = WT + (size_t)(colBase + 32 + m) * K + half * 8;
    const unsigned short* __restrict__ w3 = WT + (size_t)(colBase + 48 + m) * K + half * 8;

    for (int k0 = 0; k0 < K; k0 += 32) {
        if (k0 + 32 < K) {
            __builtin_prefetch((const void*)(ar0 + k0 + 32), 0, 1);
            __builtin_prefetch((const void*)(ar1 + k0 + 32), 0, 1);
            __builtin_prefetch((const void*)(w0 + k0 + 32), 0, 1);
            __builtin_prefetch((const void*)(w2 + k0 + 32), 0, 1);
        }
        // Load all fragments for this K step first (one long load clause),
        // then issue the 8-WMMA burst.
        BfFrag a0, a1, b0, b1, b2, b3;
        a0.h[0] = *(const u16x8*)(ar0 + k0);
        a0.h[1] = *(const u16x8*)(ar0 + k0 + 16);
        a1.h[0] = *(const u16x8*)(ar1 + k0);
        a1.h[1] = *(const u16x8*)(ar1 + k0 + 16);
        b0.h[0] = *(const u16x8*)(w0 + k0);  b0.h[1] = *(const u16x8*)(w0 + k0 + 16);
        b1.h[0] = *(const u16x8*)(w1 + k0);  b1.h[1] = *(const u16x8*)(w1 + k0 + 16);
        b2.h[0] = *(const u16x8*)(w2 + k0);  b2.h[1] = *(const u16x8*)(w2 + k0 + 16);
        b3.h[0] = *(const u16x8*)(w3 + k0);  b3.h[1] = *(const u16x8*)(w3 + k0 + 16);

        acc[0] = __builtin_amdgcn_wmma_f32_16x16x32_bf16(false, a0.bf, false, b0.bf, (short)0, acc[0], false, false);
        acc[4] = __builtin_amdgcn_wmma_f32_16x16x32_bf16(false, a1.bf, false, b0.bf, (short)0, acc[4], false, false);
        acc[1] = __builtin_amdgcn_wmma_f32_16x16x32_bf16(false, a0.bf, false, b1.bf, (short)0, acc[1], false, false);
        acc[5] = __builtin_amdgcn_wmma_f32_16x16x32_bf16(false, a1.bf, false, b1.bf, (short)0, acc[5], false, false);
        acc[2] = __builtin_amdgcn_wmma_f32_16x16x32_bf16(false, a0.bf, false, b2.bf, (short)0, acc[2], false, false);
        acc[6] = __builtin_amdgcn_wmma_f32_16x16x32_bf16(false, a1.bf, false, b2.bf, (short)0, acc[6], false, false);
        acc[3] = __builtin_amdgcn_wmma_f32_16x16x32_bf16(false, a0.bf, false, b3.bf, (short)0, acc[3], false, false);
        acc[7] = __builtin_amdgcn_wmma_f32_16x16x32_bf16(false, a1.bf, false, b3.bf, (short)0, acc[7], false, false);
    }

#pragma unroll
    for (int r = 0; r < 2; ++r) {
        const int rBase = rowBase + r * 16 + half * 8;
#pragma unroll
        for (int t = 0; t < 4; ++t) {
            const int col = colBase + t * 16 + m;
            const float bcol = bias ? bias[col] : 0.0f;
            const v8f a = acc[r * 4 + t];
#pragma unroll
            for (int j = 0; j < 8; ++j) {
                const float v = a[j] + bcol;
                const size_t o = (size_t)(rBase + j) * N + col;
                out[o] = v;
                if (out_bf) out_bf[o] = f2bf(v);
            }
        }
    }
}

// ---------------------------------------------------------------------------
// Weight transpose + f32->bf16:  WT[n*K + k] = bf16(W[k*N + n])
// ---------------------------------------------------------------------------
__global__ void k_transpose_cvt(const float* __restrict__ W,
                                unsigned short* __restrict__ WT,
                                int K, int N)
{
    int gid = blockIdx.x * blockDim.x + threadIdx.x;
    if (gid >= K * N) return;
    int n = gid / K;
    int k = gid - n * K;
    WT[gid] = f2bf(W[(size_t)k * N + n]);
}

// embedding gather -> bf16 activation matrix [ROWS x EMB]
__global__ void k_embed(const int* __restrict__ idx,
                        const float* __restrict__ emb,
                        unsigned short* __restrict__ X)
{
    int gid = blockIdx.x * blockDim.x + threadIdx.x;
    if (gid >= ROWS * EMB) return;
    int row = gid >> 7;           // / EMB
    int col = gid & (EMB - 1);
    X[gid] = f2bf(emb[idx[row] * EMB + col]);
}

// ---------------------------------------------------------------------------
// SSM scan, chunked (3 passes).  u = h_in [B, T, H] fp32.
// pass1: per (b, chunk, h) local recurrence from x=0 -> partial final state
// ---------------------------------------------------------------------------
__global__ void k_ssm_partial(const float* __restrict__ u,
                              const float* __restrict__ a_,
                              const float* __restrict__ b_,
                              float* __restrict__ pf)
{
    int gid = blockIdx.x * blockDim.x + threadIdx.x;
    if (gid >= BB * NCH * HH) return;
    int h = gid & (HH - 1);
    int c = (gid >> 9) & (NCH - 1);
    int b = gid >> 13;
    const float a = a_[h], bb = b_[h];
    const float* __restrict__ up = u + ((size_t)b * TT + (size_t)c * CS) * HH + h;
    float x = 0.0f;
#pragma unroll 4
    for (int t = 0; t < CS; ++t) x = a * x + bb * up[(size_t)t * HH];
    pf[gid] = x;
}

// pass2: per (b,h) combine the 16 chunk partials into chunk initial states
__global__ void k_ssm_prefix(const float* __restrict__ a_,
                             const float* __restrict__ pf,
                             float* __restrict__ init)
{
    int gid = blockIdx.x * blockDim.x + threadIdx.x;
    if (gid >= BB * HH) return;
    int h = gid & (HH - 1);
    int b = gid >> 9;
    const float aC = powf(a_[h], (float)CS);
    float s = 0.0f;
#pragma unroll
    for (int c = 0; c < NCH; ++c) {
        size_t o = ((size_t)b * NCH + c) * HH + h;
        init[o] = s;
        s = aC * s + pf[o];
    }
}

// pass3: re-run recurrence with correct init, write y = c*x + ob in place
__global__ void k_ssm_apply(float* __restrict__ u,
                            const float* __restrict__ a_,
                            const float* __restrict__ b_,
                            const float* __restrict__ c_,
                            const float* __restrict__ ob_,
                            const float* __restrict__ init)
{
    int gid = blockIdx.x * blockDim.x + threadIdx.x;
    if (gid >= BB * NCH * HH) return;
    int h = gid & (HH - 1);
    int c = (gid >> 9) & (NCH - 1);
    int b = gid >> 13;
    const float a = a_[h], bb = b_[h], cc = c_[h], ob = ob_[h];
    float x = init[gid];
    float* __restrict__ up = u + ((size_t)b * TT + (size_t)c * CS) * HH + h;
#pragma unroll 4
    for (int t = 0; t < CS; ++t) {
        float uu = up[(size_t)t * HH];
        x = a * x + bb * uu;
        up[(size_t)t * HH] = cc * x + ob;
    }
}

// ---------------------------------------------------------------------------
// LayerNorm(H=512) + exact GELU + residual add. One wave32 per (b,t) row.
// Also emits the updated residual as bf16 (activation input of next GEMM).
// ---------------------------------------------------------------------------
__global__ __launch_bounds__(256) void k_ln_gelu_add(
    const float* __restrict__ y,
    const float* __restrict__ s_,
    const float* __restrict__ b_,
    float* __restrict__ residual,
    unsigned short* __restrict__ res_bf)
{
    const int wave = threadIdx.x >> 5;
    const int lane = threadIdx.x & 31;
    const size_t row = (size_t)blockIdx.x * 8 + wave;
    const float* __restrict__ yr = y + row * HH;

    float v[16];
    float sum = 0.0f, sq = 0.0f;
#pragma unroll
    for (int j = 0; j < 16; ++j) {
        float t = yr[lane + j * 32];
        v[j] = t; sum += t; sq += t * t;
    }
#pragma unroll
    for (int o = 16; o >= 1; o >>= 1) {
        sum += __shfl_xor(sum, o, 32);
        sq  += __shfl_xor(sq,  o, 32);
    }
    const float mu  = sum * (1.0f / (float)HH);
    const float var = sq * (1.0f / (float)HH) - mu * mu;
    const float inv = rsqrtf(var + 1e-5f);

    float* __restrict__ rr = residual + row * HH;
    unsigned short* __restrict__ rb = res_bf + row * HH;
#pragma unroll
    for (int j = 0; j < 16; ++j) {
        const int h = lane + j * 32;
        float t = (v[j] - mu) * inv * s_[h] + b_[h];
        float g = 0.5f * t * (1.0f + erff(t * 0.70710678118654752440f));
        float nr = rr[h] + g;
        rr[h] = nr;
        rb[h] = f2bf(nr);
    }
}

// ---------------------------------------------------------------------------
// Orchestration
// ---------------------------------------------------------------------------
extern "C" void kernel_launch(void* const* d_in, const int* in_sizes, int n_in,
                              void* d_out, int out_size, void* d_ws, size_t ws_size,
                              hipStream_t stream)
{
    (void)in_sizes; (void)n_in; (void)out_size; (void)ws_size;

    const int*   idx    = (const int*)  d_in[0];
    const float* emb    = (const float*)d_in[1];
    const float* in_w   = (const float*)d_in[2];
    const float* in_b   = (const float*)d_in[3];
    const float* lin_w  = (const float*)d_in[4];
    const float* ssm_a  = (const float*)d_in[5];
    const float* ssm_b  = (const float*)d_in[6];
    const float* ssm_c  = (const float*)d_in[7];
    const float* ssm_ob = (const float*)d_in[8];
    const float* ln_s   = (const float*)d_in[9];
    const float* ln_b   = (const float*)d_in[10];
    const float* head_w = (const float*)d_in[11];
    const float* head_b = (const float*)d_in[12];
    float* logits = (float*)d_out;

    // Workspace layout (all 256B aligned)
    char* ws = (char*)d_ws;
    size_t off = 0;
    auto alloc = [&](size_t bytes) { char* p = ws + off; off += (bytes + 255) & ~(size_t)255; return p; };
    float*          residual = (float*)         alloc((size_t)ROWS * HH * 4);
    float*          h_in     = (float*)         alloc((size_t)ROWS * HH * 4);
    unsigned short* Abf      = (unsigned short*)alloc((size_t)ROWS * HH * 2);   // bf16 residual
    unsigned short* Xemb     = (unsigned short*)alloc((size_t)ROWS * EMB * 2);  // bf16 embeddings
    unsigned short* in_wT    = (unsigned short*)alloc((size_t)HH * EMB * 2);
    unsigned short* lin_wT   = (unsigned short*)alloc((size_t)LL * HH * HH * 2);
    unsigned short* head_wT  = (unsigned short*)alloc((size_t)VOCAB * HH * 2);
    float*          pf       = (float*)         alloc((size_t)BB * NCH * HH * 4);
    float*          init     = (float*)         alloc((size_t)BB * NCH * HH * 4);

    // --- Phase 0: weight prep (bf16, transposed to [N][K]) ---
    k_transpose_cvt<<<(EMB * HH + 255) / 256, 256, 0, stream>>>(in_w, in_wT, EMB, HH);
    for (int l = 0; l < LL; ++l)
        k_transpose_cvt<<<(HH * HH + 255) / 256, 256, 0, stream>>>(
            lin_w + (size_t)l * HH * HH, lin_wT + (size_t)l * HH * HH, HH, HH);
    k_transpose_cvt<<<(HH * VOCAB + 255) / 256, 256, 0, stream>>>(head_w, head_wT, HH, VOCAB);

    // --- Phase 1: embed gather -> bf16, input GEMM (writes residual f32+bf16) ---
    k_embed<<<(ROWS * EMB + 255) / 256, 256, 0, stream>>>(idx, emb, Xemb);
    k_gemm_bf16<<<dim3(ROWS / 256, HH / 64), 256, 0, stream>>>(
        Xemb, in_wT, in_b, residual, Abf, ROWS, EMB, HH);

    // --- Phase 2: layers ---
    const int scanGrid = (BB * NCH * HH + 255) / 256;
    for (int l = 0; l < LL; ++l) {
        k_gemm_bf16<<<dim3(ROWS / 256, HH / 64), 256, 0, stream>>>(
            Abf, lin_wT + (size_t)l * HH * HH, nullptr, h_in, nullptr, ROWS, HH, HH);

        k_ssm_partial<<<scanGrid, 256, 0, stream>>>(h_in, ssm_a + l * HH, ssm_b + l * HH, pf);
        k_ssm_prefix<<<(BB * HH + 255) / 256, 256, 0, stream>>>(ssm_a + l * HH, pf, init);
        k_ssm_apply<<<scanGrid, 256, 0, stream>>>(h_in, ssm_a + l * HH, ssm_b + l * HH,
                                                  ssm_c + l * HH, ssm_ob + l * HH, init);

        // LN + GELU + residual add; also refresh bf16 residual for next GEMM
        k_ln_gelu_add<<<ROWS / 8, 256, 0, stream>>>(h_in, ln_s + l * HH, ln_b + l * HH,
                                                    residual, Abf);
    }

    // --- Phase 3: head GEMM ---
    k_gemm_bf16<<<dim3(ROWS / 256, VOCAB / 64), 256, 0, stream>>>(
        Abf, head_wT, head_b, logits, nullptr, ROWS, HH, VOCAB);
}
